// DyGraphCombinedModel_31739808317572
// MI455X (gfx1250) — matrix operand order repaired and checked by reading, not compiled, not run on hardware
//
#include <hip/hip_runtime.h>
#include <math.h>

#define SEQ_LEN 20
#define USER_LEN 256
#define NTOK (SEQ_LEN * USER_LEN)   // 5120
#define NINPUT 100000
#define NCENT 64
#define NCAND 1024
#define HID 128
#define TOPK 10
#define RHO 0.02f
#define MAX_TILES (NTOK / 16 + NCENT)  // 384

typedef __attribute__((ext_vector_type(16))) __bf16 v16bf;
typedef __attribute__((ext_vector_type(8)))  float  v8f;
typedef __attribute__((ext_vector_type(2)))  float  v2f;

// ---- bf16 <-> f32 via bit ops (storage type: unsigned short) ----
__device__ __forceinline__ unsigned short f2bfbits(float f) {
    union { float f; unsigned u; } a; a.f = f;
    unsigned u = a.u;
    u += 0x7fffu + ((u >> 16) & 1u);   // round to nearest even
    return (unsigned short)(u >> 16);
}
__device__ __forceinline__ __bf16 bits2bf(unsigned short s) {
    union { unsigned short s; __bf16 b; } u; u.s = s; return u.b;
}

// =====================================================================
// Kernel A: vec_output = relu([vecs_use | te2] @ Wti1 + b) @ Wti2 + b
// FP32 WMMA (16x16x4) tiles: 8 waves x 16 rows per block (128 rows/block).
// Stores fp32 rows, bf16 rows padded K20->32, and ||v||^2.
// =====================================================================
__global__ __launch_bounds__(256) void kA_vec_mlp(
    const float* __restrict__ vecs_use, const float* __restrict__ te,
    const float* __restrict__ W1, const float* __restrict__ b1,
    const float* __restrict__ W2, const float* __restrict__ b2,
    float* __restrict__ vecF, unsigned short* __restrict__ vecBF,
    float* __restrict__ nv2) {
    __shared__ float sW1[40 * 40], sW2[40 * 20], sb1[40], sb2[20], sTe[20];
    __shared__ float inT[8][16][40];   // inputs; reused as layer-2 output
    __shared__ float h1T[8][16][40];   // relu hidden
    for (int i = threadIdx.x; i < 40 * 40; i += blockDim.x) sW1[i] = W1[i];
    for (int i = threadIdx.x; i < 40 * 20; i += blockDim.x) sW2[i] = W2[i];
    if (threadIdx.x < 40) sb1[threadIdx.x] = b1[threadIdx.x];
    if (threadIdx.x < 20) { sb2[threadIdx.x] = b2[threadIdx.x]; sTe[threadIdx.x] = te[2 * 20 + threadIdx.x]; }
    __syncthreads();

    const int wave = threadIdx.x >> 5;
    const int lane = threadIdx.x & 31;
    const int hi   = lane >> 4;        // lane half
    const int lan  = lane & 15;
    const int rowBase = blockIdx.x * 128 + wave * 16;

    // stage input tile [16 rows x 40] = [vecs_use row | te2]
    for (int i = lane; i < 16 * 40; i += 32) {
        int r = i / 40, d = i % 40;
        int row = rowBase + r;
        float vv = 0.f;
        if (row < NINPUT) vv = (d < 20) ? vecs_use[(size_t)row * 20 + d] : sTe[d - 20];
        inT[wave][r][d] = vv;
    }

    // ---- layer 1: [16x40] @ [40x40] + b1, relu ----
    for (int nt = 0; nt < 3; ++nt) {
        int N = nt * 16 + lan;
        v8f acc;
        #pragma unroll
        for (int r = 0; r < 8; ++r) acc[r] = (N < 40) ? sb1[N] : 0.f;
        #pragma unroll
        for (int ks = 0; ks < 10; ++ks) {
            int k0 = ks * 4 + hi * 2;  // f32 A 16x4: lanes 0-15 K{0,1}, lanes 16-31 K{2,3}
            v2f af, bf;
            af[0] = inT[wave][lan][k0];
            af[1] = inT[wave][lan][k0 + 1];
            bf[0] = (N < 40) ? sW1[k0 * 40 + N] : 0.f;
            bf[1] = (N < 40) ? sW1[(k0 + 1) * 40 + N] : 0.f;
            acc = __builtin_amdgcn_wmma_f32_16x16x4_f32(false, af, false, bf,
                                                        (short)0, acc, false, false);
        }
        #pragma unroll
        for (int r = 0; r < 8; ++r) {
            int M = r + hi * 8;
            if (N < 40) h1T[wave][M][N] = fmaxf(acc[r], 0.f);
        }
    }

    // ---- layer 2: [16x40] @ [40x20] + b2 ----
    for (int nt = 0; nt < 2; ++nt) {
        int N = nt * 16 + lan;
        v8f acc;
        #pragma unroll
        for (int r = 0; r < 8; ++r) acc[r] = (N < 20) ? sb2[N] : 0.f;
        #pragma unroll
        for (int ks = 0; ks < 10; ++ks) {
            int k0 = ks * 4 + hi * 2;
            v2f af, bf;
            af[0] = h1T[wave][lan][k0];
            af[1] = h1T[wave][lan][k0 + 1];
            bf[0] = (N < 20) ? sW2[k0 * 20 + N] : 0.f;
            bf[1] = (N < 20) ? sW2[(k0 + 1) * 20 + N] : 0.f;
            acc = __builtin_amdgcn_wmma_f32_16x16x4_f32(false, af, false, bf,
                                                        (short)0, acc, false, false);
        }
        #pragma unroll
        for (int r = 0; r < 8; ++r) {
            int M = r + hi * 8;
            if (N < 20) inT[wave][M][N] = acc[r];   // reuse inT as output tile
        }
    }

    // ---- epilogue: stores + norms (lane pairs (r, r^16) split d-range) ----
    {
        int r = lan;
        int row = rowBase + r;
        float pn = 0.f;
        #pragma unroll
        for (int dd = 0; dd < 10; ++dd) {
            int d = hi * 10 + dd;
            float s = inT[wave][r][d];
            pn += s * s;
            if (row < NINPUT) {
                vecF[(size_t)row * 20 + d] = s;
                vecBF[(size_t)row * 32 + d] = f2bfbits(s);
            }
        }
        pn += __shfl_xor(pn, 16, 32);
        if (row < NINPUT && hi == 0) {
            nv2[row] = pn;
            #pragma unroll
            for (int d = 20; d < 32; ++d) vecBF[(size_t)row * 32 + d] = 0;
        }
    }
}

// =====================================================================
// Kernel B: per-token 'a' = 0.5*(histMLP + timeTransferMLP) via FP32 WMMA.
// 8 waves x 16 tokens per block; 40 blocks cover 5120 tokens exactly.
// Dynamic LDS layout (floats unless noted):
//   sWs1[4000] sWo1[1600] sWs2[800] sWo2[800]
//   sbs1[40] sbo1[40] sbs2[20] sbo2[20] sTe[80]
//   segT[8][16] (int)
//   histT[8][16][100]   (hist; cols 0..19 reused for ho, 40..59 for 'a')
//   h1T  [8][16][40]    (hidden tiles, reused seq->to)
// =====================================================================
#define SMEM_B ((4000 + 1600 + 800 + 800 + 40 + 40 + 20 + 20 + 80) * 4 + 128 * 4 + \
                (8 * 16 * 100) * 4 + (8 * 16 * 40) * 4)

__global__ __launch_bounds__(256) void kB_token(
    const int* __restrict__ x, const int* __restrict__ t_slot,
    const float* __restrict__ vecs_use, const float* __restrict__ te,
    const int* __restrict__ I_array,
    const float* __restrict__ Ws1, const float* __restrict__ bs1,
    const float* __restrict__ Ws2, const float* __restrict__ bs2,
    const float* __restrict__ Wo1, const float* __restrict__ bo1,
    const float* __restrict__ Wo2, const float* __restrict__ bo2,
    float* __restrict__ aF, unsigned short* __restrict__ aBF,
    float* __restrict__ na2, int* __restrict__ cidx) {
    extern __shared__ char smemB[];
    float* sWs1 = (float*)smemB;
    float* sWo1 = sWs1 + 4000;
    float* sWs2 = sWo1 + 1600;
    float* sWo2 = sWs2 + 800;
    float* sbs1 = sWo2 + 800;
    float* sbo1 = sbs1 + 40;
    float* sbs2 = sbo1 + 40;
    float* sbo2 = sbs2 + 20;
    float* sTe  = sbo2 + 20;
    int*   segT = (int*)(sTe + 80);
    float* histB = (float*)(segT + 128);
    float* h1B   = histB + 8 * 16 * 100;

    const int tid = threadIdx.x;
    for (int i = tid; i < 4000; i += 256) sWs1[i] = Ws1[i];
    for (int i = tid; i < 1600; i += 256) sWo1[i] = Wo1[i];
    for (int i = tid; i < 800;  i += 256) sWs2[i] = Ws2[i];
    for (int i = tid; i < 800;  i += 256) sWo2[i] = Wo2[i];
    if (tid < 40) { sbs1[tid] = bs1[tid]; sbo1[tid] = bo1[tid]; }
    if (tid < 20) { sbs2[tid] = bs2[tid]; sbo2[tid] = bo2[tid]; }
    if (tid < 80) sTe[tid] = te[tid];
    __syncthreads();

    const int wave = tid >> 5;
    const int lane = tid & 31;
    const int hi   = lane >> 4;
    const int lan  = lane & 15;
    const int rowBase = blockIdx.x * 128 + wave * 16;
    float* hT = histB + wave * 16 * 100;   // [16][100]
    float* gT = h1B   + wave * 16 * 40;    // [16][40]
    int*   sg = segT  + wave * 16;

    // per-token segment
    if (lane < 16) {
        int n = rowBase + lane;
        int hh = t_slot[n] % 24;
        sg[lane] = (hh >= 22 || hh < 6) ? 0 : ((hh < 14) ? 1 : 2);
    }
    // stage history tile [16][100]: 80 (row, shift-block) pairs, 20 floats each
    for (int p = lane; p < 80; p += 32) {
        int r = p / 5, b = p % 5;
        int n = rowBase + r;
        int ss = n / USER_LEN, u = n % USER_LEN;
        int k = 4 - b;
        int sr = (ss < k) ? ss : ss - k;
        int id = x[sr * USER_LEN + u];
        const float* e = vecs_use + (size_t)id * 20;
        #pragma unroll
        for (int d = 0; d < 20; ++d) hT[r * 100 + b * 20 + d] = e[d];
    }
    // wave-local LDS dependencies from here on

    // ---- seq layer 1: [16x100] @ [100x40] + bs1, relu -> gT ----
    for (int nt = 0; nt < 3; ++nt) {
        int N = nt * 16 + lan;
        v8f acc;
        #pragma unroll
        for (int r = 0; r < 8; ++r) acc[r] = (N < 40) ? sbs1[N] : 0.f;
        #pragma unroll
        for (int ks = 0; ks < 25; ++ks) {
            int k0 = ks * 4 + hi * 2;
            v2f af, bf;
            af[0] = hT[lan * 100 + k0];
            af[1] = hT[lan * 100 + k0 + 1];
            bf[0] = (N < 40) ? sWs1[k0 * 40 + N] : 0.f;
            bf[1] = (N < 40) ? sWs1[(k0 + 1) * 40 + N] : 0.f;
            acc = __builtin_amdgcn_wmma_f32_16x16x4_f32(false, af, false, bf,
                                                        (short)0, acc, false, false);
        }
        #pragma unroll
        for (int r = 0; r < 8; ++r) {
            int M = r + hi * 8;
            if (N < 40) gT[M * 40 + N] = fmaxf(acc[r], 0.f);
        }
    }
    // ---- seq layer 2: [16x40] @ [40x20] + bs2 -> ho into hT cols 0..19 ----
    for (int nt = 0; nt < 2; ++nt) {
        int N = nt * 16 + lan;
        v8f acc;
        #pragma unroll
        for (int r = 0; r < 8; ++r) acc[r] = (N < 20) ? sbs2[N] : 0.f;
        #pragma unroll
        for (int ks = 0; ks < 10; ++ks) {
            int k0 = ks * 4 + hi * 2;
            v2f af, bf;
            af[0] = gT[lan * 40 + k0];
            af[1] = gT[lan * 40 + k0 + 1];
            bf[0] = (N < 20) ? sWs2[k0 * 20 + N] : 0.f;
            bf[1] = (N < 20) ? sWs2[(k0 + 1) * 20 + N] : 0.f;
            acc = __builtin_amdgcn_wmma_f32_16x16x4_f32(false, af, false, bf,
                                                        (short)0, acc, false, false);
        }
        #pragma unroll
        for (int r = 0; r < 8; ++r) {
            int M = r + hi * 8;
            if (N < 20) hT[M * 100 + N] = acc[r];
        }
    }
    // ---- to layer 1: xc = [hist cols 80..99 | te[seg]] @ [40x40] + bo1, relu -> gT ----
    {
        int sgl = sg[lan];
        for (int nt = 0; nt < 3; ++nt) {
            int N = nt * 16 + lan;
            v8f acc;
            #pragma unroll
            for (int r = 0; r < 8; ++r) acc[r] = (N < 40) ? sbo1[N] : 0.f;
            #pragma unroll
            for (int ks = 0; ks < 10; ++ks) {
                int k0 = ks * 4 + hi * 2;
                v2f af, bf;
                af[0] = (k0     < 20) ? hT[lan * 100 + 80 + k0]     : sTe[sgl * 20 + k0 - 20];
                af[1] = (k0 + 1 < 20) ? hT[lan * 100 + 80 + k0 + 1] : sTe[sgl * 20 + k0 + 1 - 20];
                bf[0] = (N < 40) ? sWo1[k0 * 40 + N] : 0.f;
                bf[1] = (N < 40) ? sWo1[(k0 + 1) * 40 + N] : 0.f;
                acc = __builtin_amdgcn_wmma_f32_16x16x4_f32(false, af, false, bf,
                                                            (short)0, acc, false, false);
            }
            #pragma unroll
            for (int r = 0; r < 8; ++r) {
                int M = r + hi * 8;
                if (N < 40) gT[M * 40 + N] = fmaxf(acc[r], 0.f);
            }
        }
    }
    // ---- to layer 2: [16x40] @ [40x20] + bo2; a = 0.5*(ho+xo) -> hT cols 40..59 ----
    for (int nt = 0; nt < 2; ++nt) {
        int N = nt * 16 + lan;
        v8f acc;
        #pragma unroll
        for (int r = 0; r < 8; ++r) acc[r] = (N < 20) ? sbo2[N] : 0.f;
        #pragma unroll
        for (int ks = 0; ks < 10; ++ks) {
            int k0 = ks * 4 + hi * 2;
            v2f af, bf;
            af[0] = gT[lan * 40 + k0];
            af[1] = gT[lan * 40 + k0 + 1];
            bf[0] = (N < 20) ? sWo2[k0 * 20 + N] : 0.f;
            bf[1] = (N < 20) ? sWo2[(k0 + 1) * 20 + N] : 0.f;
            acc = __builtin_amdgcn_wmma_f32_16x16x4_f32(false, af, false, bf,
                                                        (short)0, acc, false, false);
        }
        #pragma unroll
        for (int r = 0; r < 8; ++r) {
            int M = r + hi * 8;
            if (N < 20) hT[M * 100 + 40 + N] = 0.5f * (hT[M * 100 + N] + acc[r]);
        }
    }
    // ---- epilogue: stores + norms + cluster ids ----
    {
        int r = lan;
        int n = rowBase + r;
        float pn = 0.f;
        #pragma unroll
        for (int dd = 0; dd < 10; ++dd) {
            int d = hi * 10 + dd;
            float s = hT[r * 100 + 40 + d];
            pn += s * s;
            aF[(size_t)n * 20 + d] = s;
            aBF[(size_t)n * 32 + d] = f2bfbits(s);
        }
        pn += __shfl_xor(pn, 16, 32);
        if (hi == 0) {
            na2[n] = pn;
            #pragma unroll
            for (int d = 20; d < 32; ++d) aBF[(size_t)n * 32 + d] = 0;
            cidx[n] = I_array[x[n]];
        }
    }
}

// =====================================================================
// Kernel B2: bucket tokens by cluster, build 16-token tile list
// =====================================================================
__global__ __launch_bounds__(256) void kB2_bucket(
    const int* __restrict__ cidx, int* __restrict__ order,
    int* __restrict__ tileCluster, int* __restrict__ tileRow,
    int* __restrict__ tileValid, int* __restrict__ numTiles) {
    __shared__ int cnt[NCENT], base[NCENT], cur[NCENT];
    int t = threadIdx.x;
    if (t < NCENT) cnt[t] = 0;
    __syncthreads();
    for (int n = t; n < NTOK; n += blockDim.x) atomicAdd(&cnt[cidx[n]], 1);
    __syncthreads();
    if (t == 0) {
        int off = 0, nt = 0;
        for (int c = 0; c < NCENT; ++c) {
            base[c] = off; cur[c] = 0;
            int cc = cnt[c];
            int ntc = (cc + 15) / 16;
            for (int q = 0; q < ntc; ++q) {
                tileCluster[nt] = c;
                tileRow[nt] = off + q * 16;
                int rem = cc - q * 16;
                tileValid[nt] = rem < 16 ? rem : 16;
                ++nt;
            }
            off += cc;
        }
        *numTiles = nt;
    }
    __syncthreads();
    for (int n = t; n < NTOK; n += blockDim.x) {
        int c = cidx[n];
        int p = atomicAdd(&cur[c], 1);
        order[base[c] + p] = n;
    }
}

// =====================================================================
// Kernel C: per 16-token tile — WMMA bf16 dot GEMM (K=20 padded to 32).
// Selection key is -d2 (monotonic in score); exact fp32 re-score for winners.
// =====================================================================
#define SMEM_C (16 * 1024 * 4 + 16 * 32 * 2 + 16 * 4 + 16 * 4)

__global__ __launch_bounds__(256) void kC_score_topk(
    const int* __restrict__ x, const int* __restrict__ cand,
    const float* __restrict__ emb,
    const unsigned short* __restrict__ vecBF, const float* __restrict__ vecF,
    const float* __restrict__ nv2,
    const unsigned short* __restrict__ aBF, const float* __restrict__ aF,
    const float* __restrict__ na2,
    const int* __restrict__ order, const int* __restrict__ tileCluster,
    const int* __restrict__ tileRow, const int* __restrict__ tileValid,
    const int* __restrict__ numTiles, float* __restrict__ out) {
    extern __shared__ char smem[];
    float*          scoreT = (float*)smem;
    unsigned short* aTs    = (unsigned short*)(smem + 65536);
    float*          na2s   = (float*)(smem + 66560);
    int*            tokS   = (int*)(smem + 66624);

    int tile = blockIdx.x;
    if (tile >= *numTiles) return;
    int cluster = tileCluster[tile];
    int rowOff  = tileRow[tile];
    int valid   = tileValid[tile];
    int tid = threadIdx.x;

    for (int i = tid; i < 16 * 32; i += blockDim.x) {
        int r = i >> 5, kk = i & 31;
        int tok = (r < valid) ? order[rowOff + r] : -1;
        aTs[i] = (tok >= 0) ? aBF[(size_t)tok * 32 + kk] : (unsigned short)0;
    }
    if (tid < 16) {
        int tok = (tid < valid) ? order[rowOff + tid] : -1;
        tokS[tid] = tok;
        na2s[tid] = (tok >= 0) ? na2[tok] : 0.f;
    }
    __syncthreads();

    const int wave = tid >> 5;
    const int lane = tid & 31;
    const int nrow = lane & 15;
    const int hi   = lane >> 4;

    float na2r[8];
    #pragma unroll
    for (int r = 0; r < 8; ++r) na2r[r] = na2s[r + hi * 8];

    v16bf afrag;
    {
        union { uint4 q; unsigned short s[8]; } lo, hh;
        const uint4* rq = (const uint4*)(aTs + nrow * 32);
        lo.q = rq[hi];
        hh.q = rq[2 + hi];
        #pragma unroll
        for (int e = 0; e < 8; ++e) { afrag[e] = bits2bf(lo.s[e]); afrag[8 + e] = bits2bf(hh.s[e]); }
    }

    const int candBase = cluster * NCAND;
    #pragma unroll
    for (int tt = 0; tt < 8; ++tt) {
        int col = wave * 128 + tt * 16 + nrow;
        int cid = cand[candBase + col];
        v16bf bfrag;
        {
            union { uint4 q; unsigned short s[8]; } q0, q1;
            const uint4* rq = (const uint4*)(vecBF + (size_t)cid * 32);
            q0.q = rq[hi];
            q1.q = rq[2 + hi];
            #pragma unroll
            for (int e = 0; e < 8; ++e) { bfrag[e] = bits2bf(q0.s[e]); bfrag[8 + e] = bits2bf(q1.s[e]); }
        }
        v8f c = {};
        c = __builtin_amdgcn_wmma_f32_16x16x32_bf16(false, afrag, false, bfrag,
                                                    (short)0, c, false, false);
        float nvc = nv2[cid];
        #pragma unroll
        for (int r = 0; r < 8; ++r) {
            int M = r + hi * 8;
            scoreT[M * 1024 + col] = fmaf(2.f, c[r], -(na2r[r] + nvc));
        }
    }
    __syncthreads();

    for (int mi = 0; mi < 2; ++mi) {
        int M = wave * 2 + mi;
        if (M >= valid) continue;
        int tok = tokS[M];
        volatile float* row = scoreT + M * 1024;

        int selI[TOPK];
        for (int kk = 0; kk < TOPK; ++kk) {
            float bv = -INFINITY; int bi = 1 << 30;
            #pragma unroll 4
            for (int j = 0; j < 32; ++j) {
                int c0 = lane + (j << 5);
                float vv = row[c0];
                if (vv > bv || (vv == bv && c0 < bi)) { bv = vv; bi = c0; }
            }
            #pragma unroll
            for (int off = 16; off > 0; off >>= 1) {
                float ov = __shfl_xor(bv, off, 32);
                int   oi = __shfl_xor(bi, off, 32);
                if (ov > bv || (ov == bv && oi < bi)) { bv = ov; bi = oi; }
            }
            selI[kk] = bi;
            if ((bi & 31) == lane) row[bi] = -INFINITY;
        }

        float w[TOPK + 1]; int ids[TOPK + 1];
        const float* aRow = aF + (size_t)tok * 20;
        float naM = na2[tok];
        float sum = 1.0f;
        for (int kk = 0; kk < TOPK; ++kk) {
            int cid = cand[candBase + selI[kk]];
            ids[kk] = cid;
            const float* vr = vecF + (size_t)cid * 20;
            float dot = 0.f;
            #pragma unroll
            for (int d = 0; d < 20; ++d) dot += aRow[d] * vr[d];
            float d2 = naM + nv2[cid] - 2.f * dot;
            float sc = expf(-RHO * sqrtf(fmaxf(d2, 1e-12f)));
            float e = expf(sc - 1.0f);
            w[kk] = e; sum += e;
        }
        ids[TOPK] = x[tok];
        w[TOPK] = 1.0f;
        float inv = 1.0f / sum;

        for (int h = lane; h < HID; h += 32) {
            float acc = 0.f;
            #pragma unroll
            for (int kk = 0; kk <= TOPK; ++kk)
                acc += w[kk] * inv * emb[(size_t)ids[kk] * HID + h];
            out[(size_t)tok * HID + h] = acc;
        }
    }
}

// =====================================================================
extern "C" void kernel_launch(void* const* d_in, const int* in_sizes, int n_in,
                              void* d_out, int out_size, void* d_ws, size_t ws_size,
                              hipStream_t stream) {
    const int*   x       = (const int*)d_in[0];
    const int*   t_slot  = (const int*)d_in[1];
    const float* vecs    = (const float*)d_in[2];
    const float* xembnet = (const float*)d_in[3];
    const int*   I_array = (const int*)d_in[4];
    const int*   cand    = (const int*)d_in[5];
    const float* te      = (const float*)d_in[6];
    const float* Ws1 = (const float*)d_in[7];  const float* bs1 = (const float*)d_in[8];
    const float* Ws2 = (const float*)d_in[9];  const float* bs2 = (const float*)d_in[10];
    const float* Wo1 = (const float*)d_in[11]; const float* bo1 = (const float*)d_in[12];
    const float* Wo2 = (const float*)d_in[13]; const float* bo2 = (const float*)d_in[14];
    const float* Wi1 = (const float*)d_in[15]; const float* bi1 = (const float*)d_in[16];
    const float* Wi2 = (const float*)d_in[17]; const float* bi2 = (const float*)d_in[18];
    float* out = (float*)d_out;

    size_t off = 0;
    char* base = (char*)d_ws;
    auto alloc = [&](size_t bytes) -> char* {
        off = (off + 255) & ~(size_t)255;
        char* p = base + off; off += bytes; return p;
    };
    float*          vecF   = (float*)alloc((size_t)NINPUT * 20 * sizeof(float));
    unsigned short* vecBF  = (unsigned short*)alloc((size_t)NINPUT * 32 * sizeof(unsigned short));
    float*          nv2    = (float*)alloc((size_t)NINPUT * sizeof(float));
    float*          aF     = (float*)alloc((size_t)NTOK * 20 * sizeof(float));
    unsigned short* aBF    = (unsigned short*)alloc((size_t)NTOK * 32 * sizeof(unsigned short));
    float*          na2    = (float*)alloc((size_t)NTOK * sizeof(float));
    int*            cidx   = (int*)alloc((size_t)NTOK * sizeof(int));
    int*            order  = (int*)alloc((size_t)NTOK * sizeof(int));
    int*            tClus  = (int*)alloc((size_t)MAX_TILES * sizeof(int));
    int*            tRow   = (int*)alloc((size_t)MAX_TILES * sizeof(int));
    int*            tValid = (int*)alloc((size_t)MAX_TILES * sizeof(int));
    int*            nTiles = (int*)alloc(sizeof(int));
    (void)ws_size; (void)in_sizes; (void)n_in; (void)out_size;

    kA_vec_mlp<<<(NINPUT + 127) / 128, 256, 0, stream>>>(
        vecs, te, Wi1, bi1, Wi2, bi2, vecF, vecBF, nv2);

    kB_token<<<NTOK / 128, 256, SMEM_B, stream>>>(
        x, t_slot, vecs, te, I_array,
        Ws1, bs1, Ws2, bs2, Wo1, bo1, Wo2, bo2,
        aF, aBF, na2, cidx);

    kB2_bucket<<<1, 256, 0, stream>>>(cidx, order, tClus, tRow, tValid, nTiles);

    kC_score_topk<<<MAX_TILES, 256, SMEM_C, stream>>>(
        x, cand, xembnet, vecBF, vecF, nv2, aBF, aF, na2,
        order, tClus, tRow, tValid, nTiles, out);
}
